// LocalLatentAttention_89275190215322
// MI455X (gfx1250) — matrix operand order repaired and checked by reading, not compiled, not run on hardware
//
#include <hip/hip_runtime.h>
#include <hip/hip_bf16.h>

#define B_    4
#define S_    8192
#define D_    512
#define H_    8
#define DH_   64
#define WIN_  128
#define CTOK_ 64      // latent tokens (chunks)
#define CHSZ_ 128     // chunk size = S/CTOK
#define LD_   256
#define LH_   4
#define LDH_  64
#define NT_   (B_*S_) // 32768 tokens

typedef __attribute__((ext_vector_type(16))) __bf16 v16bf;
typedef __attribute__((ext_vector_type(8)))  float  v8f;
typedef __attribute__((ext_vector_type(16))) unsigned short v16u;
typedef __attribute__((ext_vector_type(8)))  unsigned short v8u;

// -------- CDNA5 async global->LDS path (inline asm; builtin signature in this
// toolchain is descriptor-style, so we emit the VGLOBAL instruction directly).
#if defined(__has_builtin)
#if __has_builtin(__builtin_amdgcn_s_wait_asynccnt)
#define USE_ASYNC_LDS 1
#endif
#endif
#ifndef USE_ASYNC_LDS
#define USE_ASYNC_LDS 0
#endif

// copy 8 ushorts (16B) global -> LDS
__device__ __forceinline__ void copy8u(unsigned short* dst, const unsigned short* src) {
#if USE_ASYNC_LDS
  // LDS generic address low 32 bits == AS(3) offset (ISA aperture rule)
  unsigned lds_off = (unsigned)(unsigned long long)dst;
  asm volatile("global_load_async_to_lds_b128 %0, %1, off"
               :: "v"(lds_off), "v"(src)
               : "memory");
#else
  *(v8u*)dst = *(const v8u*)src;
#endif
}

__device__ __forceinline__ void async_join() {
#if USE_ASYNC_LDS
  __builtin_amdgcn_s_wait_asynccnt(0);
#endif
}

// ---------------------------------------------------------------------------
__device__ __forceinline__ unsigned short f2bf(float f) {
  unsigned u = __builtin_bit_cast(unsigned, f);
  u += 0x7FFFu + ((u >> 16) & 1u);           // round-to-nearest-even
  return (unsigned short)(u >> 16);
}

__device__ __forceinline__ v8f vzero() {
  v8f v = {0.f,0.f,0.f,0.f,0.f,0.f,0.f,0.f};
  return v;
}

// A fragment: 16x32 bf16 tile, row-major LDS [row][k], row = lane&15.
__device__ __forceinline__ v16bf frag_a(const unsigned short* base, int stride, int kbase) {
  const int lane = threadIdx.x & 31;
  const int row  = lane & 15;
  const int k0   = kbase + ((lane >> 4) << 3);
  const unsigned short* p = base + row * stride + k0;
  v8u lo = *(const v8u*)(p);
  v8u hi = *(const v8u*)(p + 16);
  v16u v;
#pragma unroll
  for (int i = 0; i < 8; ++i) { v[i] = lo[i]; v[i + 8] = hi[i]; }
  return __builtin_bit_cast(v16bf, v);
}

// B fragment: 32x16 bf16, LDS stored [n][k], n = lane&15, K split 8+8 (table order).
__device__ __forceinline__ v16bf frag_b(const unsigned short* base, int stride, int kbase) {
  const int lane = threadIdx.x & 31;
  const int n  = lane & 15;
  const int k0 = kbase + ((lane >> 4) << 3);
  const unsigned short* p = base + n * stride + k0;
  v8u lo = *(const v8u*)(p);
  v8u hi = *(const v8u*)(p + 16);
  v16u v;
#pragma unroll
  for (int i = 0; i < 8; ++i) { v[i] = lo[i]; v[i + 8] = hi[i]; }
  return __builtin_bit_cast(v16bf, v);
}

__device__ __forceinline__ v8f wmma_bf16(v16bf a, v16bf b, v8f c) {
  return __builtin_amdgcn_wmma_f32_16x16x32_bf16(false, a, false, b, (short)0, c, false, false);
}

// ---------------------------------------------------------------------------
// f32 -> bf16 elementwise (8 elems/thread)
// ---------------------------------------------------------------------------
__global__ __launch_bounds__(256) void convert_bf16(const float* __restrict__ src,
                                                    unsigned short* __restrict__ dst)
{
  size_t g = ((size_t)blockIdx.x * 256 + threadIdx.x) * 8;
  float4 f0 = *(const float4*)(src + g);
  float4 f1 = *(const float4*)(src + g + 4);
  v8u o;
  o[0]=f2bf(f0.x); o[1]=f2bf(f0.y); o[2]=f2bf(f0.z); o[3]=f2bf(f0.w);
  o[4]=f2bf(f1.x); o[5]=f2bf(f1.y); o[6]=f2bf(f1.z); o[7]=f2bf(f1.w);
  *(v8u*)(dst + g) = o;
}

// W[K][N] f32 -> Wt[N][K] bf16 (grid = N)
__global__ __launch_bounds__(256) void transpose_w(const float* __restrict__ W,
                                                   unsigned short* __restrict__ Wt,
                                                   int K, int N)
{
  int n = blockIdx.x;
  for (int k = threadIdx.x; k < K; k += 256)
    Wt[(size_t)n * K + k] = f2bf(W[(size_t)k * N + n]);
}

// ---------------------------------------------------------------------------
// Double-buffered bf16 GEMM: C[M,N] = A_bf[M,K] @ Wt_bf[N,K]^T + bias.
// grid = (N/128, M/128), block = 256 (8 waves).
// ---------------------------------------------------------------------------
template <bool BF16OUT>
__global__ __launch_bounds__(256) void gemm_tile(
    const unsigned short* __restrict__ Ab,   // [M][K]
    const unsigned short* __restrict__ Wt,   // [N][K]
    const float* __restrict__ bias,
    void* __restrict__ out, int ldc, int K)
{
  __shared__ unsigned short AsBuf[2 * 128 * 48];
  __shared__ unsigned short WsBuf[2 * 128 * 48];
  const int tid  = threadIdx.x;
  const int wave = tid >> 5, lane = tid & 31;
  const int wm = wave & 3, wn = wave >> 2;
  const int hi = lane >> 4, nn = lane & 15;
  const int row0 = blockIdx.y * 128, col0 = blockIdx.x * 128;

  v8f acc[2][4];
#pragma unroll
  for (int i = 0; i < 2; ++i)
#pragma unroll
    for (int j = 0; j < 4; ++j) acc[i][j] = vzero();

  const int r = tid >> 2, c8 = (tid & 3) * 8;        // 128 rows x 4 chunks
  const int r2 = r + 64;                              // second half

  // issue tile kb into buffer buf (each thread: 2 A-chunks + 2 W-chunks)
  auto issue = [&](int kb, int buf) {
    unsigned short* As = AsBuf + buf * (128 * 48);
    unsigned short* Ws = WsBuf + buf * (128 * 48);
    copy8u(As + r  * 48 + c8, Ab + (size_t)(row0 + r)  * K + kb + c8);
    copy8u(As + r2 * 48 + c8, Ab + (size_t)(row0 + r2) * K + kb + c8);
    copy8u(Ws + r  * 48 + c8, Wt + (size_t)(col0 + r)  * K + kb + c8);
    copy8u(Ws + r2 * 48 + c8, Wt + (size_t)(col0 + r2) * K + kb + c8);
  };

  issue(0, 0);
  int buf = 0;
  for (int kb = 0; kb < K; kb += 32) {
    async_join();
    __syncthreads();
    if (kb + 32 < K) issue(kb + 32, buf ^ 1);
    const unsigned short* As = AsBuf + buf * (128 * 48);
    const unsigned short* Ws = WsBuf + buf * (128 * 48);
    v16bf a0 = frag_a(As + (wm * 32 + 0)  * 48, 48, 0);
    v16bf a1 = frag_a(As + (wm * 32 + 16) * 48, 48, 0);
#pragma unroll
    for (int ni = 0; ni < 4; ++ni) {
      v16bf bv = frag_b(Ws + (wn * 64 + ni * 16) * 48, 48, 0);
      acc[0][ni] = wmma_bf16(a0, bv, acc[0][ni]);
      acc[1][ni] = wmma_bf16(a1, bv, acc[1][ni]);
    }
    buf ^= 1;
  }
#pragma unroll
  for (int mi = 0; mi < 2; ++mi)
#pragma unroll
    for (int ni = 0; ni < 4; ++ni)
#pragma unroll
      for (int j = 0; j < 8; ++j) {
        int m = wm * 32 + mi * 16 + j + hi * 8;
        int n = wn * 64 + ni * 16 + nn;
        float v = acc[mi][ni][j] + bias[col0 + n];
        size_t idx = (size_t)(row0 + m) * ldc + col0 + n;
        if constexpr (BF16OUT) ((unsigned short*)out)[idx] = f2bf(v);
        else                   ((float*)out)[idx] = v;
      }
}

// ---------------------------------------------------------------------------
// Chunk mean pooling: pooled_bf[b,c,:] = mean over 128 tokens (bf16 out)
// ---------------------------------------------------------------------------
__global__ __launch_bounds__(256) void pool_kernel(const float* __restrict__ x,
                                                   unsigned short* __restrict__ pooled)
{
  int blk = blockIdx.x;         // B*CTOK
  int b = blk >> 6, c = blk & 63;
  const float* xp = x + ((size_t)b * S_ + (size_t)c * CHSZ_) * D_;
  for (int d = threadIdx.x; d < D_; d += 256) {
    float s = 0.f;
    for (int j = 0; j < CHSZ_; ++j) s += xp[(size_t)j * D_ + d];
    pooled[(size_t)blk * D_ + d] = f2bf(s * (1.0f / CHSZ_));
  }
}

// ---------------------------------------------------------------------------
// Local sliding-window flash attention: 64-query tiles, up to 3 key tiles of
// 64, online softmax via LDS score round-trip, bf16 WMMA throughout.
// grid = B*H*(S/64), block = 256.
// ---------------------------------------------------------------------------
__global__ __launch_bounds__(256) void local_attn(
    const unsigned short* __restrict__ Qb,
    const unsigned short* __restrict__ Kb,
    const unsigned short* __restrict__ Vb,
    float* __restrict__ Out)
{
  __shared__ unsigned short Qs[64 * 72];
  __shared__ unsigned short Ks[64 * 72];
  __shared__ unsigned short Vt[64 * 72];   // [d][key]
  __shared__ float Sc[64 * 68];            // scores; aliased as bf16 P in-place
  __shared__ float mst[64], lst[64], alp[64];

  const int tid  = threadIdx.x;
  const int wave = tid >> 5, lane = tid & 31;
  const int wm = wave & 3, wn = wave >> 2;
  const int hi = lane >> 4, nn = lane & 15;

  const int nTiles = S_ / 64;
  int bid = blockIdx.x;
  int t = bid & (nTiles - 1);
  int h = (bid / nTiles) & (H_ - 1);
  int b = bid / (nTiles * H_);
  const size_t bS = (size_t)b * S_;
  const int qbase = t * 64;

  { // Q tile 64x64 (async)
    int r = tid >> 2, seg = (tid & 3) * 16;
    const unsigned short* src = Qb + (bS + qbase + r) * D_ + h * DH_ + seg;
    copy8u(Qs + r * 72 + seg,     src);
    copy8u(Qs + r * 72 + seg + 8, src + 8);
  }
  async_join();
  if (tid < 64) { mst[tid] = -1e30f; lst[tid] = 0.f; }

  v8f o0 = vzero(), o1 = vzero();

  int kt0 = t - 2; if (kt0 < 0) kt0 = 0;
  for (int kt = kt0; kt <= t; ++kt) {
    const int kb0 = kt * 64;
    __syncthreads();
    { // K tile [key][d] (async) + V tile transposed [d][key] (sync scatter)
      int j = tid >> 2, seg = (tid & 3) * 16;
      const unsigned short* vsrc = Vb + (bS + kb0 + j) * D_ + h * DH_ + seg;
      v8u b0 = *(const v8u*)(vsrc);
      v8u b1 = *(const v8u*)(vsrc + 8);
#pragma unroll
      for (int i = 0; i < 8; ++i) {
        Vt[(seg + i)     * 72 + j] = b0[i];
        Vt[(seg + 8 + i) * 72 + j] = b1[i];
      }
      const unsigned short* ksrc = Kb + (bS + kb0 + j) * D_ + h * DH_ + seg;
      copy8u(Ks + j * 72 + seg,     ksrc);
      copy8u(Ks + j * 72 + seg + 8, ksrc + 8);
    }
    async_join();
    __syncthreads();
    // S = Q @ K^T : wave computes rows wm*16..+15, cols wn*32..+31
    {
      v16bf a0 = frag_a(Qs + (wm * 16) * 72, 72, 0);
      v16bf a1 = frag_a(Qs + (wm * 16) * 72, 72, 32);
#pragma unroll
      for (int ni = 0; ni < 2; ++ni) {
        v8f s = vzero();
        v16bf bv0 = frag_b(Ks + (wn * 32 + ni * 16) * 72, 72, 0);
        v16bf bv1 = frag_b(Ks + (wn * 32 + ni * 16) * 72, 72, 32);
        s = wmma_bf16(a0, bv0, s);
        s = wmma_bf16(a1, bv1, s);
#pragma unroll
        for (int j = 0; j < 8; ++j)
          Sc[(wm * 16 + j + hi * 8) * 68 + wn * 32 + ni * 16 + nn] = s[j];
      }
    }
    __syncthreads();
    // online softmax; thread r owns full row r. P written bf16 in-place.
    if (tid < 64) {
      const int r = tid;
      const int qpos = qbase + r;
      const float scale = 0.125f;   // 1/sqrt(64)
      float mprev = mst[r], lprev = lst[r];
      float mt = -1e30f;
      for (int j = 0; j < 64; ++j) {
        int dist = qpos - (kb0 + j);
        if (dist >= 0 && dist < WIN_) mt = fmaxf(mt, Sc[r * 68 + j] * scale);
      }
      float mnew = fmaxf(mprev, mt);
      float alpha = __expf(mprev - mnew);
      float psum = 0.f;
      unsigned short* Pu = (unsigned short*)Sc;
      for (int j = 0; j < 64; ++j) {
        int dist = qpos - (kb0 + j);
        float p = 0.f;
        if (dist >= 0 && dist < WIN_) p = __expf(Sc[r * 68 + j] * scale - mnew);
        psum += p;
        Pu[r * 136 + j] = f2bf(p);   // safe: ascending in-place alias
      }
      mst[r] = mnew; lst[r] = alpha * lprev + psum; alp[r] = alpha;
    }
    __syncthreads();
    // rescale accumulators, then O += P @ V
    {
      const unsigned short* Pu = (const unsigned short*)Sc;
#pragma unroll
      for (int j = 0; j < 8; ++j) {
        float a_ = alp[wm * 16 + j + hi * 8];
        o0[j] *= a_; o1[j] *= a_;
      }
      v16bf pa0 = frag_a(Pu + (wm * 16) * 136, 136, 0);
      v16bf pa1 = frag_a(Pu + (wm * 16) * 136, 136, 32);
#pragma unroll
      for (int ni = 0; ni < 2; ++ni) {
        v16bf bv0 = frag_b(Vt + (wn * 32 + ni * 16) * 72, 72, 0);
        v16bf bv1 = frag_b(Vt + (wn * 32 + ni * 16) * 72, 72, 32);
        v8f* op = ni ? &o1 : &o0;
        *op = wmma_bf16(pa0, bv0, *op);
        *op = wmma_bf16(pa1, bv1, *op);
      }
    }
  }
  __syncthreads();
#pragma unroll
  for (int ni = 0; ni < 2; ++ni)
#pragma unroll
    for (int j = 0; j < 8; ++j) {
      int r = wm * 16 + j + hi * 8;
      float inv = 1.0f / lst[r];
      int c = wn * 32 + ni * 16 + nn;
      float val = (ni ? o1[j] : o0[j]) * inv;
      Out[(bS + qbase + r) * D_ + h * DH_ + c] = val;
    }
}

// ---------------------------------------------------------------------------
// Latent attention: 64 queries x 64 latents, chunk-causal mask, one tile.
// Writes bf16 output (feeds remote GEMM). grid = B*LH*(S/64), block = 256.
// ---------------------------------------------------------------------------
__global__ __launch_bounds__(256) void latent_attn(
    const unsigned short* __restrict__ RQ,   // [NT][256]
    const unsigned short* __restrict__ GK,   // [B*64][256]
    const unsigned short* __restrict__ GV,
    unsigned short* __restrict__ LatB)       // [NT][256] bf16
{
  __shared__ unsigned short Qs[64 * 72];
  __shared__ unsigned short Gk[64 * 72];
  __shared__ unsigned short Gt[64 * 72];     // [d][latent]
  __shared__ float Sc[64 * 68];

  const int tid  = threadIdx.x;
  const int wave = tid >> 5, lane = tid & 31;
  const int wm = wave & 3, wn = wave >> 2;
  const int hi = lane >> 4, nn = lane & 15;

  const int nTiles = S_ / 64;
  int bid = blockIdx.x;
  int t  = bid & (nTiles - 1);
  int lh = (bid / nTiles) & (LH_ - 1);
  int b  = bid / (nTiles * LH_);
  const size_t bS = (size_t)b * S_;
  const int qbase = t * 64;

  {
    int r = tid >> 2, seg = (tid & 3) * 16;
    const unsigned short* src = RQ + (bS + qbase + r) * LD_ + lh * LDH_ + seg;
    copy8u(Qs + r * 72 + seg,     src);
    copy8u(Qs + r * 72 + seg + 8, src + 8);
    const unsigned short* gks = GK + ((size_t)b * CTOK_ + r) * LD_ + lh * LDH_ + seg;
    copy8u(Gk + r * 72 + seg,     gks);
    copy8u(Gk + r * 72 + seg + 8, gks + 8);
    const unsigned short* gvs = GV + ((size_t)b * CTOK_ + r) * LD_ + lh * LDH_ + seg;
    v8u b0 = *(const v8u*)(gvs);
    v8u b1 = *(const v8u*)(gvs + 8);
#pragma unroll
    for (int i = 0; i < 8; ++i) {
      Gt[(seg + i)     * 72 + r] = b0[i];
      Gt[(seg + 8 + i) * 72 + r] = b1[i];
    }
  }
  async_join();
  __syncthreads();
  {
    v16bf a0 = frag_a(Qs + (wm * 16) * 72, 72, 0);
    v16bf a1 = frag_a(Qs + (wm * 16) * 72, 72, 32);
#pragma unroll
    for (int ni = 0; ni < 2; ++ni) {
      v8f s = vzero();
      v16bf bv0 = frag_b(Gk + (wn * 32 + ni * 16) * 72, 72, 0);
      v16bf bv1 = frag_b(Gk + (wn * 32 + ni * 16) * 72, 72, 32);
      s = wmma_bf16(a0, bv0, s);
      s = wmma_bf16(a1, bv1, s);
#pragma unroll
      for (int j = 0; j < 8; ++j)
        Sc[(wm * 16 + j + hi * 8) * 68 + wn * 32 + ni * 16 + nn] = s[j];
    }
  }
  __syncthreads();
  if (tid < 64) {
    const int r = tid;
    const int qpos = qbase + r;
    const int cid = qpos >> 7;          // allowed latents: c < cid
    unsigned short* Pu = (unsigned short*)Sc;
    const float scale = 0.125f;
    if (cid == 0) {
      for (int j = 0; j < 64; ++j) Pu[r * 136 + j] = 0;
    } else {
      int lim = cid < 64 ? cid : 64;
      float m = -1e30f;
      for (int j = 0; j < lim; ++j) m = fmaxf(m, Sc[r * 68 + j] * scale);
      float l = 0.f;
      for (int j = 0; j < 64; ++j) {
        float p = 0.f;
        if (j < lim) p = __expf(Sc[r * 68 + j] * scale - m);
        l += p;
        Sc[r * 68 + j] = p;
      }
      float inv = 1.0f / l;
      for (int j = 0; j < 64; ++j) Pu[r * 136 + j] = f2bf(Sc[r * 68 + j] * inv);
    }
  }
  __syncthreads();
  v8f o0 = vzero(), o1 = vzero();
  {
    const unsigned short* Pu = (const unsigned short*)Sc;
    v16bf pa0 = frag_a(Pu + (wm * 16) * 136, 136, 0);
    v16bf pa1 = frag_a(Pu + (wm * 16) * 136, 136, 32);
#pragma unroll
    for (int ni = 0; ni < 2; ++ni) {
      v16bf bv0 = frag_b(Gt + (wn * 32 + ni * 16) * 72, 72, 0);
      v16bf bv1 = frag_b(Gt + (wn * 32 + ni * 16) * 72, 72, 32);
      v8f* op = ni ? &o1 : &o0;
      *op = wmma_bf16(pa0, bv0, *op);
      *op = wmma_bf16(pa1, bv1, *op);
    }
  }
#pragma unroll
  for (int ni = 0; ni < 2; ++ni)
#pragma unroll
    for (int j = 0; j < 8; ++j) {
      int r = wm * 16 + j + hi * 8;
      int c = wn * 32 + ni * 16 + nn;
      LatB[(bS + qbase + r) * LD_ + lh * LDH_ + c] = f2bf(ni ? o1[j] : o0[j]);
    }
}

// ---------------------------------------------------------------------------
// Gate + fuse: gate = sigmoid([x, local] @ Wg + bg); fused (bf16) written.
// One wave per token; grid = NT/8, block = 256.
// ---------------------------------------------------------------------------
__global__ __launch_bounds__(256) void gate_fuse(
    const float* __restrict__ x,
    const float* __restrict__ local,
    const float* __restrict__ remote,
    const float* __restrict__ Wg, const float* __restrict__ bg,
    unsigned short* __restrict__ fusedB)
{
  const int wave = threadIdx.x >> 5, lane = threadIdx.x & 31;
  const size_t tk = (size_t)blockIdx.x * 8 + wave;
  const float* xp = x + tk * D_;
  const float* lp = local + tk * D_;
  const float* rp = remote + tk * D_;
  unsigned short* fp = fusedB + tk * D_;
  float z = 0.f;
  for (int i = lane; i < D_; i += 32)
    z += xp[i] * Wg[i] + lp[i] * Wg[D_ + i];
#pragma unroll
  for (int off = 16; off > 0; off >>= 1)
    z += __shfl_xor(z, off, 32);
  float g = 1.0f / (1.0f + __expf(-(z + bg[0])));
  for (int i = lane; i < D_; i += 32)
    fp[i] = f2bf(g * lp[i] + (1.0f - g) * rp[i]);
}

// ---------------------------------------------------------------------------
extern "C" void kernel_launch(void* const* d_in, const int* in_sizes, int n_in,
                              void* d_out, int out_size, void* d_ws, size_t ws_size,
                              hipStream_t stream) {
  (void)in_sizes; (void)n_in; (void)out_size; (void)ws_size;
  const float* x   = (const float*)d_in[0];
  const float* Wq  = (const float*)d_in[1];  const float* bq  = (const float*)d_in[2];
  const float* Wk  = (const float*)d_in[3];  const float* bk  = (const float*)d_in[4];
  const float* Wv  = (const float*)d_in[5];  const float* bv  = (const float*)d_in[6];
  const float* Wrq = (const float*)d_in[7];  const float* brq = (const float*)d_in[8];
  const float* Wrk = (const float*)d_in[9];  const float* brk = (const float*)d_in[10];
  const float* Wrv = (const float*)d_in[11]; const float* brv = (const float*)d_in[12];
  const float* Wro = (const float*)d_in[13]; const float* bro = (const float*)d_in[14];
  const float* Wo  = (const float*)d_in[15]; const float* bo  = (const float*)d_in[16];
  const float* Wg  = (const float*)d_in[17]; const float* bg  = (const float*)d_in[18];

  char* ws = (char*)d_ws;
  size_t off = 0;
  auto take = [&](size_t bytes) -> char* {
    char* p = ws + off;
    off = (off + bytes + 255) & ~(size_t)255;
    return p;
  };
  unsigned short* x_bf   = (unsigned short*)take((size_t)NT_ * D_ * 2);
  unsigned short* wqT    = (unsigned short*)take((size_t)D_ * D_ * 2);
  unsigned short* wkT    = (unsigned short*)take((size_t)D_ * D_ * 2);
  unsigned short* wvT    = (unsigned short*)take((size_t)D_ * D_ * 2);
  unsigned short* woT    = (unsigned short*)take((size_t)D_ * D_ * 2);
  unsigned short* wrqT   = (unsigned short*)take((size_t)LD_ * D_ * 2);
  unsigned short* wrkT   = (unsigned short*)take((size_t)LD_ * D_ * 2);
  unsigned short* wrvT   = (unsigned short*)take((size_t)LD_ * D_ * 2);
  unsigned short* wroT   = (unsigned short*)take((size_t)D_ * LD_ * 2);
  unsigned short* q_bf   = (unsigned short*)take((size_t)NT_ * D_ * 2);
  unsigned short* k_bf   = (unsigned short*)take((size_t)NT_ * D_ * 2);
  unsigned short* v_bf   = (unsigned short*)take((size_t)NT_ * D_ * 2);
  unsigned short* rq_bf  = (unsigned short*)take((size_t)NT_ * LD_ * 2);
  unsigned short* pooled = (unsigned short*)take((size_t)B_ * CTOK_ * D_ * 2);
  unsigned short* gk_bf  = (unsigned short*)take((size_t)B_ * CTOK_ * LD_ * 2);
  unsigned short* gv_bf  = (unsigned short*)take((size_t)B_ * CTOK_ * LD_ * 2);
  float*          local  = (float*)take((size_t)NT_ * D_ * 4);
  unsigned short* lat_bf = (unsigned short*)take((size_t)NT_ * LD_ * 2);
  float*          remote = (float*)take((size_t)NT_ * D_ * 4);
  unsigned short* fus_bf = (unsigned short*)take((size_t)NT_ * D_ * 2);

  dim3 blk(256);
  // one-time operand conversion
  convert_bf16<<<(NT_ * D_) / (256 * 8), blk, 0, stream>>>(x, x_bf);
  transpose_w<<<D_,  blk, 0, stream>>>(Wq,  wqT,  D_,  D_);
  transpose_w<<<D_,  blk, 0, stream>>>(Wk,  wkT,  D_,  D_);
  transpose_w<<<D_,  blk, 0, stream>>>(Wv,  wvT,  D_,  D_);
  transpose_w<<<D_,  blk, 0, stream>>>(Wo,  woT,  D_,  D_);
  transpose_w<<<LD_, blk, 0, stream>>>(Wrq, wrqT, D_,  LD_);
  transpose_w<<<LD_, blk, 0, stream>>>(Wrk, wrkT, D_,  LD_);
  transpose_w<<<LD_, blk, 0, stream>>>(Wrv, wrvT, D_,  LD_);
  transpose_w<<<D_,  blk, 0, stream>>>(Wro, wroT, LD_, D_);
  // projections (bf16 outputs)
  gemm_tile<true><<<dim3(D_/128,  NT_/128), blk, 0, stream>>>(x_bf, wqT,  bq,  q_bf,  D_,  D_);
  gemm_tile<true><<<dim3(D_/128,  NT_/128), blk, 0, stream>>>(x_bf, wkT,  bk,  k_bf,  D_,  D_);
  gemm_tile<true><<<dim3(D_/128,  NT_/128), blk, 0, stream>>>(x_bf, wvT,  bv,  v_bf,  D_,  D_);
  gemm_tile<true><<<dim3(LD_/128, NT_/128), blk, 0, stream>>>(x_bf, wrqT, brq, rq_bf, LD_, D_);
  // latent K/V from pooled chunks
  pool_kernel<<<B_*CTOK_, blk, 0, stream>>>(x, pooled);
  gemm_tile<true><<<dim3(LD_/128, (B_*CTOK_)/128), blk, 0, stream>>>(pooled, wrkT, brk, gk_bf, LD_, D_);
  gemm_tile<true><<<dim3(LD_/128, (B_*CTOK_)/128), blk, 0, stream>>>(pooled, wrvT, brv, gv_bf, LD_, D_);
  // attention
  local_attn<<<B_*H_*(S_/64),   blk, 0, stream>>>(q_bf, k_bf, v_bf, local);
  latent_attn<<<B_*LH_*(S_/64), blk, 0, stream>>>(rq_bf, gk_bf, gv_bf, lat_bf);
  // remote projection (f32 out)
  gemm_tile<false><<<dim3(D_/128, NT_/128), blk, 0, stream>>>(lat_bf, wroT, bro, remote, D_, LD_);
  // gate + fuse -> bf16 fused
  gate_fuse<<<NT_/8, blk, 0, stream>>>(x, local, remote, Wg, bg, fus_bf);
  // output projection (f32 out to d_out)
  gemm_tile<false><<<dim3(D_/128, NT_/128), blk, 0, stream>>>(fus_bf, woT, bo, d_out, D_, D_);
}
